// OrderInvariantKernelLinearAttention_8650064134242
// MI455X (gfx1250) — matrix operand order repaired and checked by reading, not compiled, not run on hardware
//
#include <hip/hip_runtime.h>

// Hedgehog linear attention forward, chunked scan.
// B=2, H=32, L=4096, D=128, F=64, C=32, Df=128, N=128 chunks.
// One workgroup (256 threads = 8 wave32) per (b,h); S held in WMMA accumulators.

#define C_BLK   32
#define D_DIM   128
#define F_DIM   64
#define DF_DIM  128
#define L_SEQ   4096
#define N_BLK   (L_SEQ / C_BLK)
#define CHUNK_ELEMS (C_BLK * D_DIM)   // 4096
#define EPS_F   1e-6f

typedef __bf16 bf16;
typedef __attribute__((ext_vector_type(4)))  bf16  bf16x4;
typedef __attribute__((ext_vector_type(8)))  bf16  bf16x8;
typedef __attribute__((ext_vector_type(16))) bf16  bf16x16;
typedef __attribute__((ext_vector_type(8)))  float f32x8;

// ---- LDS layout (bytes) ----
#define OFF_QBF   0        // 32x128 bf16        (8192)
#define OFF_KBF   8192     // 32x128 bf16        (8192)
#define OFF_VT    16384    // 128x32 bf16  vT[d][c]   (8192)
#define OFF_WT    24576    // 64x128 bf16  WT[f][d]   (16384)
#define OFF_ST    40960    // 128x128 bf16 ST[d][f]   (32768)
#define OFF_U     73728    // 2 x (32x64) f32    (16384)
#define OFF_PHIQ  90112    // 32x128 bf16 phi_q[c][f] (8192)
#define OFF_PHIKT 98304    // 128x32 bf16 phi_kT[f][c](8192)
#define OFF_Z     106496   // 128 f32           (512)
#define OFF_RDEN  107008   // 32 f32            (128)
#define SHMEM_BYTES 107136

// A-operand fragment, 16-bit data, 16x32 tile, row-major source (row stride applied
// by caller). Per ISA layout: lane half selects K groups {half*8..+7, 16+half*8..+7}.
__device__ __forceinline__ bf16x16 load_fragA(const bf16* row_ptr, int half) {
  bf16x8 lo = *(const bf16x8*)(row_ptr + half * 8);
  bf16x8 hi = *(const bf16x8*)(row_ptr + 16 + half * 8);
  return __builtin_shufflevector(lo, hi, 0,1,2,3,4,5,6,7,8,9,10,11,12,13,14,15);
}

// B-operand fragment, 16-bit data, 32x16 tile. Source is B^T row-major (column of B
// contiguous along K). Lane half selects K groups {half*8..+7, 16+half*8..+7}.
__device__ __forceinline__ bf16x16 load_fragB(const bf16* col_ptr, int half) {
  bf16x8 lo = *(const bf16x8*)(col_ptr + half * 8);
  bf16x8 hi = *(const bf16x8*)(col_ptr + 16 + half * 8);
  return __builtin_shufflevector(lo, hi, 0,1,2,3,4,5,6,7,8,9,10,11,12,13,14,15);
}

__device__ __forceinline__ f32x8 wmma_bf16(bf16x16 a, bf16x16 b, f32x8 c) {
  return __builtin_amdgcn_wmma_f32_16x16x32_bf16(false, a, false, b, (short)0, c,
                                                 false, false);
}

extern "C" __global__ __launch_bounds__(256)
void hedgehog_linattn_kernel(const float* __restrict__ qg,
                             const float* __restrict__ kg,
                             const float* __restrict__ vg,
                             const float* __restrict__ Wg,
                             float* __restrict__ outg,
                             int H) {
  extern __shared__ char smem[];
  bf16*  q_bf   = (bf16*)(smem + OFF_QBF);
  bf16*  k_bf   = (bf16*)(smem + OFF_KBF);
  bf16*  vT     = (bf16*)(smem + OFF_VT);
  bf16*  WT     = (bf16*)(smem + OFF_WT);
  bf16*  ST     = (bf16*)(smem + OFF_ST);
  float* ubuf   = (float*)(smem + OFF_U);
  bf16*  phi_q  = (bf16*)(smem + OFF_PHIQ);
  bf16*  phi_kT = (bf16*)(smem + OFF_PHIKT);
  float* Zv     = (float*)(smem + OFF_Z);
  float* rden   = (float*)(smem + OFF_RDEN);

  const int tid  = threadIdx.x;
  const int lane = tid & 31;
  const int w    = tid >> 5;       // wave id 0..7
  const int half = lane >> 4;      // lane-half for WMMA layouts
  const int ln   = lane & 15;      // row/col within tile

  const int bh = blockIdx.x;       // b*H + h
  const int h  = bh % H;
  const size_t seq_base = (size_t)bh * L_SEQ * D_DIM;

  // Load W^T (per-head constant) once: WT[f][d] = W[h][d][f]
  const float* Wh = Wg + (size_t)h * D_DIM * F_DIM;
  for (int idx = tid; idx < D_DIM * F_DIM; idx += 256) {
    int d = idx >> 6, f = idx & 63;
    WT[f * D_DIM + d] = (bf16)Wh[idx];
  }
  if (tid < DF_DIM) Zv[tid] = 0.0f;

  // Running state S (Df x D) in registers: wave w owns f-tile-row w, 8 d-tiles.
  f32x8 S[8];
#pragma unroll
  for (int i = 0; i < 8; ++i)
    S[i] = (f32x8){0.f, 0.f, 0.f, 0.f, 0.f, 0.f, 0.f, 0.f};

  __syncthreads();

  for (int n = 0; n < N_BLK; ++n) {
    const size_t blk = seq_base + (size_t)n * CHUNK_ELEMS;

    // ---- load q,k,v chunk with 128-bit loads; v stored transposed vT[d][c] ----
#pragma unroll
    for (int it = 0; it < CHUNK_ELEMS / (256 * 4); ++it) {
      int idx4 = (tid + it * 256) * 4;          // 4 consecutive floats per thread
      int c = idx4 >> 7, d = idx4 & 127;
      float4 qv = *(const float4*)(qg + blk + idx4);
      float4 kv = *(const float4*)(kg + blk + idx4);
      float4 vv = *(const float4*)(vg + blk + idx4);
      bf16x4 qb = {(bf16)qv.x, (bf16)qv.y, (bf16)qv.z, (bf16)qv.w};
      bf16x4 kb = {(bf16)kv.x, (bf16)kv.y, (bf16)kv.z, (bf16)kv.w};
      *(bf16x4*)(q_bf + idx4) = qb;
      *(bf16x4*)(k_bf + idx4) = kb;
      vT[(d + 0) * C_BLK + c] = (bf16)vv.x;
      vT[(d + 1) * C_BLK + c] = (bf16)vv.y;
      vT[(d + 2) * C_BLK + c] = (bf16)vv.z;
      vT[(d + 3) * C_BLK + c] = (bf16)vv.w;
    }

    // ---- prefetch next chunk (sequential scan => perfectly predictable) ----
    if (n + 1 < N_BLK) {
      const size_t nb = blk + CHUNK_ELEMS + (size_t)tid * 16;  // 64B per thread
      __builtin_prefetch(qg + nb, 0, 1);
      __builtin_prefetch(kg + nb, 0, 1);
      __builtin_prefetch(vg + nb, 0, 1);
    }
    __syncthreads();

    // ---- u = x @ W for x in {q,k}: 16 tiles of 16x16, 2 per wave, K=128 ----
#pragma unroll
    for (int j = 0; j < 2; ++j) {
      int id = w * 2 + j;
      const bf16* src = (id < 8) ? q_bf : k_bf;
      float* udst = ubuf + ((id < 8) ? 0 : (C_BLK * F_DIM));
      int t = id & 7, mt = t >> 2, nt = t & 3;
      f32x8 acc = {0.f, 0.f, 0.f, 0.f, 0.f, 0.f, 0.f, 0.f};
#pragma unroll
      for (int kb = 0; kb < 4; ++kb) {
        bf16x16 a = load_fragA(src + (mt * 16 + ln) * D_DIM + kb * 32, half);
        bf16x16 b = load_fragB(WT + (nt * 16 + ln) * D_DIM + kb * 32, half);
        acc = wmma_bf16(a, b, acc);
      }
#pragma unroll
      for (int i = 0; i < 8; ++i)
        udst[(mt * 16 + half * 8 + i) * F_DIM + nt * 16 + ln] = acc[i];
    }
    __syncthreads();

    // ---- feature map: phi = [softmax(u), softmax(-u)] per row ----
    if (tid < 64) {
      int mat = tid >> 5, c = tid & 31;
      const float* ur = ubuf + mat * (C_BLK * F_DIM) + c * F_DIM;
      float mx = -1e30f, mn = 1e30f;
      for (int f = 0; f < F_DIM; ++f) {
        float x = ur[f];
        mx = fmaxf(mx, x);
        mn = fminf(mn, x);
      }
      float s1 = 0.f, s2 = 0.f;
      for (int f = 0; f < F_DIM; ++f) {
        s1 += __expf(ur[f] - mx);
        s2 += __expf(mn - ur[f]);
      }
      float r1 = 1.f / s1, r2 = 1.f / s2;
      if (mat == 0) {
        for (int f = 0; f < F_DIM; ++f) {
          phi_q[c * DF_DIM + f]         = (bf16)(__expf(ur[f] - mx) * r1);
          phi_q[c * DF_DIM + F_DIM + f] = (bf16)(__expf(mn - ur[f]) * r2);
        }
      } else {
        for (int f = 0; f < F_DIM; ++f) {
          phi_kT[f * C_BLK + c]           = (bf16)(__expf(ur[f] - mx) * r1);
          phi_kT[(F_DIM + f) * C_BLK + c] = (bf16)(__expf(mn - ur[f]) * r2);
        }
      }
    }
    __syncthreads();

    // ---- S += phi_k^T @ v (register accumulators), write bf16 shadow ST ----
    {
      bf16x16 a = load_fragA(phi_kT + (w * 16 + ln) * C_BLK, half);
#pragma unroll
      for (int dt = 0; dt < 8; ++dt) {
        bf16x16 b = load_fragB(vT + (dt * 16 + ln) * C_BLK, half);
        S[dt] = wmma_bf16(a, b, S[dt]);
      }
#pragma unroll
      for (int dt = 0; dt < 8; ++dt) {
        alignas(16) bf16 tmp[8];
#pragma unroll
        for (int i = 0; i < 8; ++i) tmp[i] = (bf16)S[dt][i];
        *(bf16x8*)(ST + (dt * 16 + ln) * DF_DIM + w * 16 + half * 8) =
            *(const bf16x8*)tmp;
      }
    }
    // ---- Z[f] += sum_c phi_k[c][f] ----
    if (tid < DF_DIM) {
      float s = 0.f;
      for (int c = 0; c < C_BLK; ++c) s += (float)phi_kT[tid * C_BLK + c];
      Zv[tid] += s;
    }
    __syncthreads();

    // ---- den[c] = max(phi_q[c,:] . Z, eps); store reciprocal (wave 0) ----
    if (tid < C_BLK) {
      float s = 0.f;
      for (int f = 0; f < DF_DIM; ++f)
        s += (float)phi_q[tid * DF_DIM + f] * Zv[f];
      rden[tid] = 1.0f / fmaxf(s, EPS_F);
    }

    // ---- num = phi_q @ S : 16 tiles, 2 per wave, K=128 ----
    f32x8 acc2[2];
#pragma unroll
    for (int j = 0; j < 2; ++j) {
      int id = w * 2 + j, mt = id >> 3, dt = id & 7;
      f32x8 acc = {0.f, 0.f, 0.f, 0.f, 0.f, 0.f, 0.f, 0.f};
#pragma unroll
      for (int kb = 0; kb < 4; ++kb) {
        bf16x16 a = load_fragA(phi_q + (mt * 16 + ln) * DF_DIM + kb * 32, half);
        bf16x16 b = load_fragB(ST + (dt * 16 + ln) * DF_DIM + kb * 32, half);
        acc = wmma_bf16(a, b, acc);
      }
      acc2[j] = acc;
    }
    __syncthreads();  // rden visible to all; also quiesce before next-iter loads

    // ---- scale by 1/den and store output ----
#pragma unroll
    for (int j = 0; j < 2; ++j) {
      int id = w * 2 + j, mt = id >> 3, dt = id & 7;
      const int c0 = mt * 16 + half * 8;
#pragma unroll
      for (int i = 0; i < 8; ++i) {
        float r = rden[c0 + i];
        outg[blk + (size_t)(c0 + i) * D_DIM + dt * 16 + ln] = acc2[j][i] * r;
      }
    }
    __syncthreads();
  }
}

extern "C" void kernel_launch(void* const* d_in, const int* in_sizes, int n_in,
                              void* d_out, int out_size, void* d_ws, size_t ws_size,
                              hipStream_t stream) {
  const float* q = (const float*)d_in[0];
  const float* k = (const float*)d_in[1];
  const float* v = (const float*)d_in[2];
  const float* W = (const float*)d_in[3];
  float* out = (float*)d_out;

  const int BH = in_sizes[0] / (L_SEQ * D_DIM);              // B*H = 64
  const int H  = in_sizes[3] / (D_DIM * F_DIM);              // 32

  (void)hipFuncSetAttribute((const void*)hedgehog_linattn_kernel,
                            hipFuncAttributeMaxDynamicSharedMemorySize,
                            (int)SHMEM_BYTES);

  hipLaunchKernelGGL(hedgehog_linattn_kernel, dim3(BH), dim3(256), SHMEM_BYTES,
                     stream, q, k, v, W, out, H);
}
